// ScaleDotProductAttention_34978213658631
// MI455X (gfx1250) — matrix operand order repaired and checked by reading, not compiled, not run on hardware
//
#include <hip/hip_runtime.h>

// ---------------------------------------------------------------------------
// Scaled-dot-product attention for MI455X (gfx1250, wave32, WMMA).
// B=4 H=16 S=2048 D=64, fp32 in/out. d_out = attended [B,H,S,D] ++ weights
// [B,H,S,S].
//
// Roofline: the mandatory 1.07GB fp32 weight write dominates (~47us at
// 23.3TB/s); compute (~69 GFLOP) is negligible in f16 WMMA. Design:
//  * prep kernel: K -> f16 (Kh, same layout), V -> f16 transposed (Vt[d][key])
//    into d_ws so all WMMA operand loads are contiguous b128s (L2-resident).
//  * main kernel per 16-query tile (4 waves): QK^T WMMA -> scores f16 in LDS
//    (66KB of the 320KB WGP LDS); per-LANE max only (no shuffles) + LDS table
//    scan for row max; single exp2 per element; PV WMMA from LDS A-fragments;
//    1/l folded into the weight/output stores.
//  * nontemporal hints: weight/O stores and read-once loads bypass L2
//    retention so the reused f16 workspace + mask stay cached.
// ---------------------------------------------------------------------------

typedef _Float16 h16 __attribute__((ext_vector_type(16)));
typedef _Float16 h8  __attribute__((ext_vector_type(8)));
typedef _Float16 h4  __attribute__((ext_vector_type(4)));
typedef float    f8  __attribute__((ext_vector_type(8)));
typedef float    f4  __attribute__((ext_vector_type(4)));

#if defined(__has_builtin)
#if __has_builtin(__builtin_amdgcn_exp2f)
#define EXP2F(x) __builtin_amdgcn_exp2f(x)
#endif
#endif
#ifndef EXP2F
#define EXP2F(x) exp2f(x)
#endif

#define LOG2E 1.4426950408889634f

constexpr int S_    = 2048;
constexpr int D_    = 64;
constexpr int TQ    = 16;          // query rows per workgroup
constexpr int NWAVE = 4;           // waves per workgroup (128 threads)
constexpr int PITCH = 2048 + 16;   // f16 score-row pitch (bank padding)
constexpr int BH_   = 64;          // B*H

// LDS layout (dynamic) for the main kernel
constexpr int S16_BYTES  = TQ * PITCH * 2;            // 66048
constexpr int MPITCH     = 68;                        // 64 entries + pad
constexpr int STATM_OFF  = S16_BYTES;                 // float[16][68]
constexpr int LPITCH     = 12;                        // 8 entries + pad (16B align)
constexpr int STATL_OFF  = STATM_OFF + 16 * MPITCH * 4;
constexpr int SMEM_MAIN  = STATL_OFF + 16 * LPITCH * 4;

// ---------------------------------------------------------------------------
// Prep: Kh[bh][key][d] = f16(K), Vt[bh][d][key] = f16(V) (transposed).
// One block = one (bh, 64-key tile). 256 threads.
// ---------------------------------------------------------------------------
__global__ __launch_bounds__(256)
void attn_prep_kernel(const float* __restrict__ K, const float* __restrict__ V,
                      _Float16* __restrict__ Kh, _Float16* __restrict__ Vt)
{
  __shared__ _Float16 tile[64][72];
  const int bh = blockIdx.y;
  const int kt = blockIdx.x;            // 64-key tile
  const int t  = threadIdx.x;

  const float* Kb = K + ((size_t)bh * S_ + kt * 64) * D_;
  const float* Vb = V + ((size_t)bh * S_ + kt * 64) * D_;
  _Float16*   Khb = Kh + ((size_t)bh * S_ + kt * 64) * D_;

  const int row = t >> 4;               // 0..15
  const int c4  = (t & 15) * 4;         // 0..60
#pragma unroll
  for (int i = 0; i < 4; ++i) {
    const int r = row + i * 16;
    // K: straight f32 -> f16 convert; f32 side is read-once -> NT load
    f4 xk = __builtin_nontemporal_load((const f4*)(Kb + (size_t)r * D_ + c4));
    h4 yk;
#pragma unroll
    for (int j = 0; j < 4; ++j) yk[j] = (_Float16)xk[j];
    *(h4*)(Khb + (size_t)r * D_ + c4) = yk;     // re-read by main: keep RT
    // V: stage into LDS for transpose
    f4 xv = __builtin_nontemporal_load((const f4*)(Vb + (size_t)r * D_ + c4));
#pragma unroll
    for (int j = 0; j < 4; ++j) tile[r][c4 + j] = (_Float16)xv[j];
  }
  __syncthreads();
  {
    const int d  = t >> 2;              // 0..63
    const int ks = (t & 3) * 16;        // 0,16,32,48
    h8 a, b;
#pragma unroll
    for (int j = 0; j < 8; ++j) { a[j] = tile[ks + j][d]; b[j] = tile[ks + 8 + j][d]; }
    _Float16* dst = Vt + ((size_t)bh * D_ + d) * S_ + kt * 64 + ks;
    *(h8*)dst       = a;                        // re-read by main: keep RT
    *(h8*)(dst + 8) = b;
  }
}

// ---------------------------------------------------------------------------
// Main attention kernel. PREP: use f16 Kh/Vt from workspace (fast path).
// ---------------------------------------------------------------------------
template <bool PREP>
__global__ __launch_bounds__(128)
void attn_main_kernel(const float* __restrict__ Q, const float* __restrict__ Kf,
                      const float* __restrict__ Vf, const float* __restrict__ Mk,
                      const _Float16* __restrict__ Kh, const _Float16* __restrict__ Vt,
                      float* __restrict__ Out, float* __restrict__ W)
{
  extern __shared__ char smem_raw[];
  _Float16* s16    = (_Float16*)smem_raw;                 // [TQ][PITCH] scores / p~
  float*    stat_m = (float*)(smem_raw + STATM_OFF);      // [16][MPITCH]
  float*    stat_l = (float*)(smem_raw + STATL_OFF);      // [16][LPITCH]

  const int lane  = threadIdx.x & 31;
  const int wave  = threadIdx.x >> 5;
  const int lo16  = lane & 15;
  const bool hi   = lane >= 16;
  const int qbase = blockIdx.x * TQ;
  const int bh    = blockIdx.y;

  const float* Qb = Q  + (size_t)bh * S_ * D_;
  float*       Ob = Out + (size_t)bh * S_ * D_;
  float*       Wb = W   + (size_t)bh * S_ * S_;

  // ---- Q A-fragments (16x32 f16), d-chunks {0,32} -------------------------
  // A layout: lane l holds row l%16; lane<16: K = d0+{0..7,16..23},
  //           lane>=16: K = d0+{8..15,24..31}.
  h16 qa[2];
  {
    const float* qrow = Qb + (size_t)(qbase + lo16) * D_;
    const int doff = hi ? 8 : 0;
#pragma unroll
    for (int c = 0; c < 2; ++c) {
      const int d0 = c * 32 + doff;
      f4 x0 = __builtin_nontemporal_load((const f4*)(qrow + d0));       // Q read once
      f4 x1 = __builtin_nontemporal_load((const f4*)(qrow + d0 + 4));
      f4 x2 = __builtin_nontemporal_load((const f4*)(qrow + d0 + 16));
      f4 x3 = __builtin_nontemporal_load((const f4*)(qrow + d0 + 20));
#pragma unroll
      for (int i = 0; i < 4; ++i) {
        qa[c][i]      = (_Float16)x0[i];
        qa[c][4 + i]  = (_Float16)x1[i];
        qa[c][8 + i]  = (_Float16)x2[i];
        qa[c][12 + i] = (_Float16)x3[i];
      }
    }
  }

  // ---- Pass 1: QK^T WMMA; scores (log2-domain) -> LDS f16; per-LANE max ---
  float m_lane[8];
#pragma unroll
  for (int e = 0; e < 8; ++e) m_lane[e] = -3.0e38f;
  const float k1 = 0.125f * LOG2E;     // 1/sqrt(64) * log2(e)

  for (int t = wave; t < S_ / 16; t += NWAVE) {
    const int kbase = t * 16;
    f8 acc = {};
#pragma unroll
    for (int c = 0; c < 2; ++c) {
      // B layout (32x16): lane n holds col n%16; K(d) = {0..7,16..23} (+8 for hi).
      h16 kb;
      if constexpr (PREP) {
        const _Float16* kr = Kh + (size_t)(kbase + lo16) * D_ + c * 32 + (hi ? 8 : 0);
        h8 k0 = *(const h8*)kr;
        h8 k1v = *(const h8*)(kr + 16);
        kb = __builtin_shufflevector(k0, k1v, 0,1,2,3,4,5,6,7,8,9,10,11,12,13,14,15);
      } else {
        const float* kr = Kf + (size_t)bh * S_ * D_ + (size_t)(kbase + lo16) * D_
                        + c * 32 + (hi ? 8 : 0);
        f4 y0 = *(const f4*)(kr);
        f4 y1 = *(const f4*)(kr + 4);
        f4 y2 = *(const f4*)(kr + 16);
        f4 y3 = *(const f4*)(kr + 20);
#pragma unroll
        for (int i = 0; i < 4; ++i) {
          kb[i] = (_Float16)y0[i];      kb[4 + i]  = (_Float16)y1[i];
          kb[8 + i] = (_Float16)y2[i];  kb[12 + i] = (_Float16)y3[i];
        }
      }
      acc = __builtin_amdgcn_wmma_f32_16x16x32_f16(
          false, qa[c], false, kb, (short)0, acc, false, false);
    }
    // C layout: element e -> row e + hi*8, col lo16.
#pragma unroll
    for (int e = 0; e < 8; ++e) {
      const int row = e + (hi ? 8 : 0);
      const float mv = Mk[(size_t)(qbase + row) * S_ + kbase + lo16];  // L2-resident
      const float sv = acc[e] * k1 + mv * LOG2E;
      m_lane[e] = fmaxf(m_lane[e], sv);
      s16[row * PITCH + kbase + lo16] = (_Float16)sv;
    }
  }

  // per-lane maxima -> LDS table [row][wave*16 + col]
#pragma unroll
  for (int e = 0; e < 8; ++e) {
    const int row = e + (hi ? 8 : 0);
    stat_m[row * MPITCH + wave * 16 + lo16] = m_lane[e];
  }
  __syncthreads();

  // row max for this lane's A-layout row (scan 64 entries, vectorized)
  const int myrow = lo16;
  float m_f = -3.0e38f;
#pragma unroll
  for (int i = 0; i < 16; ++i) {
    f4 v = *(const f4*)&stat_m[myrow * MPITCH + i * 4];
    m_f = fmaxf(m_f, fmaxf(fmaxf(v[0], v[1]), fmaxf(v[2], v[3])));
  }

  // ---- Pass 2a: p~ = exp2(s - m) -> LDS f16; per-lane sum -----------------
  {
    _Float16* srow = s16 + myrow * PITCH;
    float lsum = 0.0f;
    for (int c = wave; c < S_ / 32; c += NWAVE) {
      const int kc = c * 32 + (hi ? 8 : 0);
      h8 sl = *(h8*)(srow + kc);         // A-frag elems 0..7  (k = kc..kc+7)
      h8 sh = *(h8*)(srow + kc + 16);    // A-frag elems 8..15 (k = kc+16..+23)
      h8 p0, p1;
#pragma unroll
      for (int i = 0; i < 8; ++i) {
        float pa = EXP2F((float)sl[i] - m_f);
        float pb = EXP2F((float)sh[i] - m_f);
        lsum += pa + pb;
        p0[i] = (_Float16)pa;
        p1[i] = (_Float16)pb;
      }
      *(h8*)(srow + kc)      = p0;
      *(h8*)(srow + kc + 16) = p1;
    }
    stat_l[myrow * LPITCH + wave * 2 + (hi ? 1 : 0)] = lsum;
  }
  __syncthreads();

  // 1/l for this lane's row (weights) and for the 8 output rows (O scaling)
  float invl;
  {
    f4 a = *(const f4*)&stat_l[myrow * LPITCH];
    f4 b = *(const f4*)&stat_l[myrow * LPITCH + 4];
    invl = 1.0f / (a[0] + a[1] + a[2] + a[3] + b[0] + b[1] + b[2] + b[3]);
  }
  float invO[8];
#pragma unroll
  for (int e = 0; e < 8; ++e) {
    const int row = e + (hi ? 8 : 0);
    f4 a = *(const f4*)&stat_l[row * LPITCH];
    f4 b = *(const f4*)&stat_l[row * LPITCH + 4];
    invO[e] = 1.0f / (a[0] + a[1] + a[2] + a[3] + b[0] + b[1] + b[2] + b[3]);
  }

  // ---- Pass 2b: O = (P~ x V) * invl ; fused NT fp32 weight writes ---------
  {
    const int dcol = wave * 16 + lo16;
    _Float16* srow = s16 + myrow * PITCH;
    float* wrow = Wb + (size_t)(qbase + myrow) * S_;
    f8 acc = {};
    for (int c = 0; c < S_ / 32; ++c) {
      const int kc = c * 32 + (hi ? 8 : 0);
      h8 a0 = *(h8*)(srow + kc);
      h8 a1 = *(h8*)(srow + kc + 16);
      h16 a = __builtin_shufflevector(a0, a1, 0,1,2,3,4,5,6,7,8,9,10,11,12,13,14,15);
      h16 b;
      if constexpr (PREP) {
        const _Float16* vr = Vt + ((size_t)bh * D_ + dcol) * S_ + c * 32 + (hi ? 8 : 0);
        h8 v0 = *(const h8*)vr;
        h8 v1 = *(const h8*)(vr + 16);
        b = __builtin_shufflevector(v0, v1, 0,1,2,3,4,5,6,7,8,9,10,11,12,13,14,15);
      } else {
        const float* vcol = Vf + (size_t)bh * S_ * D_
                          + (size_t)(c * 32 + (hi ? 8 : 0)) * D_ + dcol;
#pragma unroll
        for (int i = 0; i < 16; ++i) b[i] = (_Float16)vcol[(size_t)(i < 8 ? i : i + 8) * D_];
      }
      acc = __builtin_amdgcn_wmma_f32_16x16x32_f16(
          false, a, false, b, (short)0, acc, false, false);
      if ((c & (NWAVE - 1)) == wave) {   // wave-uniform: this wave owns chunk c
        f4 w0 = {(float)a0[0]*invl, (float)a0[1]*invl, (float)a0[2]*invl, (float)a0[3]*invl};
        f4 w1 = {(float)a0[4]*invl, (float)a0[5]*invl, (float)a0[6]*invl, (float)a0[7]*invl};
        f4 w2 = {(float)a1[0]*invl, (float)a1[1]*invl, (float)a1[2]*invl, (float)a1[3]*invl};
        f4 w3 = {(float)a1[4]*invl, (float)a1[5]*invl, (float)a1[6]*invl, (float)a1[7]*invl};
        // 1.07GB write-once stream: NT so it doesn't sweep L2
        __builtin_nontemporal_store(w0, (f4*)(wrow + kc));
        __builtin_nontemporal_store(w1, (f4*)(wrow + kc + 4));
        __builtin_nontemporal_store(w2, (f4*)(wrow + kc + 16));
        __builtin_nontemporal_store(w3, (f4*)(wrow + kc + 20));
      }
    }
#pragma unroll
    for (int e = 0; e < 8; ++e) {
      const int row = e + (hi ? 8 : 0);
      __builtin_nontemporal_store(acc[e] * invO[e],
                                  Ob + (size_t)(qbase + row) * D_ + dcol);
    }
  }
}

extern "C" void kernel_launch(void* const* d_in, const int* in_sizes, int n_in,
                              void* d_out, int out_size, void* d_ws, size_t ws_size,
                              hipStream_t stream) {
  (void)in_sizes; (void)n_in; (void)out_size;
  const float* q  = (const float*)d_in[0];
  const float* k  = (const float*)d_in[1];
  const float* v  = (const float*)d_in[2];
  const float* mk = (const float*)d_in[3];
  float* out = (float*)d_out;                         // attended [B,H,S,D]
  float* w   = out + (size_t)BH_ * S_ * D_;           // weights  [B,H,S,S]

  const size_t half_elems = (size_t)BH_ * S_ * D_;    // per tensor
  const size_t need = 2 * half_elems * sizeof(_Float16);
  dim3 grid(S_ / TQ, BH_);
  dim3 block(32 * NWAVE);

  if (d_ws != nullptr && ws_size >= need) {
    _Float16* kh = (_Float16*)d_ws;
    _Float16* vt = kh + half_elems;
    attn_prep_kernel<<<dim3(S_ / 64, BH_), 256, 0, stream>>>(k, v, kh, vt);
    attn_main_kernel<true><<<grid, block, SMEM_MAIN, stream>>>(
        q, k, v, mk, kh, vt, out, w);
  } else {
    attn_main_kernel<false><<<grid, block, SMEM_MAIN, stream>>>(
        q, k, v, mk, nullptr, nullptr, out, w);
  }
}
